// InteractionBlock_7275674599645
// MI455X (gfx1250) — matrix operand order repaired and checked by reading, not compiled, not run on hardware
//
#include <hip/hip_runtime.h>
#include <hip/hip_bf16.h>

typedef __bf16 bf16;
typedef __attribute__((ext_vector_type(16))) __bf16 v16bf;
typedef __attribute__((ext_vector_type(8)))  float  v8f;
typedef __attribute__((ext_vector_type(4)))  int    v4i;

#define NN 50000
#define EE 600000
#define DD 128
#define HH 8
#define CC 16
#define DFF 512

#define EDGE_NT 4          // tiles (16 edges each) per block in k_edge
#define LDSPAD 4           // fp32 row padding -> stride 132 dwords (132 mod 64 = 4, conflict-free frags)

#if __has_builtin(__builtin_amdgcn_global_load_async_to_lds_b128) && \
    __has_builtin(__builtin_amdgcn_s_wait_asynccnt)
#define HAS_ASYNC_LDS 1
typedef __attribute__((address_space(1))) v4i gv4i;   // global-AS 16B vector
typedef __attribute__((address_space(3))) v4i lv4i;   // LDS-AS 16B vector
#else
#define HAS_ASYNC_LDS 0
#endif

// ---------- helpers ----------
__device__ __forceinline__ unsigned enc_f32(float f) {
    unsigned u = __float_as_uint(f);
    return (u & 0x80000000u) ? ~u : (u | 0x80000000u);
}
__device__ __forceinline__ float dec_f32(unsigned u) {
    unsigned v = (u & 0x80000000u) ? (u & 0x7FFFFFFFu) : ~u;
    return __uint_as_float(v);
}

// A fragment (16x32 bf16, MxK): lane = {half, m}; lanes 0-15 hold K {k0..k0+7, k0+16..k0+23}
__device__ __forceinline__ v16bf load_a_frag_g(const bf16* A, size_t row0, int ldk, int k0, int lane) {
    int half = lane >> 4, m = lane & 15;
    const bf16* ar = A + (row0 + (size_t)m) * (size_t)ldk + k0;
    v16bf a;
#pragma unroll
    for (int j = 0; j < 8; ++j) {
        int k = (j < 4) ? (half * 8 + 2 * j) : (half * 8 + 16 + 2 * (j - 4));
        a[2 * j]     = ar[k];
        a[2 * j + 1] = ar[k + 1];
    }
    return a;
}

// B fragment (32x16 bf16, KxN) from Wt stored [Nout][K]: lanes 0-15 hold K {k0..k0+7, k0+16..k0+23}, lanes 16-31 {k0+8..k0+15, k0+24..k0+31}
__device__ __forceinline__ v16bf load_b_frag(const bf16* Wt, int col0, int K, int k0, int lane) {
    int half = lane >> 4, n = lane & 15;
    const bf16* br = Wt + (size_t)(col0 + n) * (size_t)K + k0 + half * 8;
    v16bf b;
#pragma unroll
    for (int j = 0; j < 8; ++j) { int k = (j < 4) ? 2 * j : 16 + 2 * (j - 4); b[2 * j] = br[k]; b[2 * j + 1] = br[k + 1]; }
    return b;
}

// ---------- weight cast + transpose: W[K][Nout] f32 -> Wt[Nout][K] bf16 ----------
__global__ void k_wt(const float* __restrict__ W, bf16* __restrict__ Wt, int K, int Nout) {
    long idx = (long)blockIdx.x * blockDim.x + threadIdx.x;
    if (idx >= (long)K * Nout) return;
    int k = (int)(idx / Nout), n = (int)(idx % Nout);
    Wt[(size_t)n * K + k] = (bf16)W[idx];
}

// ---------- init scratch ----------
__global__ void k_init(float* __restrict__ aggr, float* __restrict__ z, unsigned* __restrict__ mmax) {
    long idx = (long)blockIdx.x * blockDim.x + threadIdx.x;
    if (idx < (long)NN * DD) aggr[idx] = 0.f;
    if (idx < (long)NN * HH) { z[idx] = 0.f; mmax[idx] = 0x007FFFFFu; /* enc(-inf) */ }
}

// ---------- LayerNorm (+ optional residual combine) ----------
__global__ void k_ln(const float* __restrict__ xin, const float* __restrict__ aggr,
                     const float* __restrict__ skip, const float* __restrict__ alphap,
                     const float* __restrict__ g, const float* __restrict__ beta,
                     float* __restrict__ x2out, bf16* __restrict__ hout) {
    int lane = threadIdx.x & 31;
    int row  = blockIdx.x * 8 + (threadIdx.x >> 5);
    size_t base = (size_t)row * DD + lane * 4;
    float4 v = *(const float4*)(xin + base);
    if (aggr) {
        float a = alphap[0];
        float4 s = *(const float4*)(skip + base);
        float4 m = *(const float4*)(aggr + base);
        v.x += a * (m.x + s.x); v.y += a * (m.y + s.y);
        v.z += a * (m.z + s.z); v.w += a * (m.w + s.w);
    }
    if (x2out) *(float4*)(x2out + base) = v;
    float sum = v.x + v.y + v.z + v.w;
    float sq  = v.x * v.x + v.y * v.y + v.z * v.z + v.w * v.w;
#pragma unroll
    for (int off = 16; off >= 1; off >>= 1) {
        sum += __shfl_xor(sum, off, 32);
        sq  += __shfl_xor(sq,  off, 32);
    }
    float mean = sum * (1.f / DD);
    float var  = sq * (1.f / DD) - mean * mean;
    float rs   = rsqrtf(var + 1e-5f);
    int c = lane * 4;
    bf16* ho = hout + base;
    ho[0] = (bf16)((v.x - mean) * rs * g[c + 0] + beta[c + 0]);
    ho[1] = (bf16)((v.y - mean) * rs * g[c + 1] + beta[c + 1]);
    ho[2] = (bf16)((v.z - mean) * rs * g[c + 2] + beta[c + 2]);
    ho[3] = (bf16)((v.w - mean) * rs * g[c + 3] + beta[c + 3]);
}

// ---------- generic bf16 WMMA GEMM: C[M,Nout] = A[M,K] @ Wt^T + bias ----------
// mode 0: outF = acc+bias ; mode 1: outB = silu(acc+bias) bf16 ; mode 2: outF = base + acc + bias
__global__ void k_gemm(const bf16* __restrict__ A, const bf16* __restrict__ Wt,
                       const float* __restrict__ bias, const float* __restrict__ base,
                       float* __restrict__ outF, bf16* __restrict__ outB,
                       int K, int Nout, int mode) {
    int w = threadIdx.x >> 5, lane = threadIdx.x & 31;
    size_t row0 = (size_t)blockIdx.x * 16;
    int col0 = blockIdx.y * 128 + w * 16;
    v8f acc = {};
    for (int k0 = 0; k0 < K; k0 += 32) {
        if (k0 + 32 < K)
            __builtin_prefetch(A + (row0 + (lane & 15)) * (size_t)K + k0 + 32, 0, 1);
        v16bf a = load_a_frag_g(A, row0, K, k0, lane);
        v16bf b = load_b_frag(Wt, col0, K, k0, lane);
        acc = __builtin_amdgcn_wmma_f32_16x16x32_bf16(false, a, false, b, (short)0, acc, false, false);
    }
    int half = lane >> 4, n = lane & 15;
    int col = col0 + n;
    float bia = bias ? bias[col] : 0.f;
#pragma unroll
    for (int r = 0; r < 8; ++r) {
        size_t row = row0 + r + 8 * half;
        size_t idx = row * (size_t)Nout + col;
        float v = acc[r] + bia;
        if (mode == 0) {
            outF[idx] = v;
        } else if (mode == 1) {
            outB[idx] = (bf16)(v / (1.f + __expf(-v)));
        } else {
            outF[idx] = base[idx] + v;
        }
    }
}

// ---------- fused edge kernel ----------
// Per block: EDGE_NT tiles of 16 edges. Tile pipeline:
//   async(global fp32 -> LDS) double-buffered, WMMA e = edge_attr@We,
//   store e (bf16), per-head logits via cross-lane reduce, atomic segment-max.
__global__ void k_edge(const float* __restrict__ edge_attr, const long long* __restrict__ ei,
                       const bf16* __restrict__ WeT, const float* __restrict__ Qn,
                       const float* __restrict__ Kn, bf16* __restrict__ e_bf,
                       float* __restrict__ logits, unsigned* __restrict__ mmax) {
    __shared__ float sF[2][16][DD + LDSPAD];
    int t = threadIdx.x;
    int w = t >> 5, lane = t & 31;
    int half = lane >> 4, nl = lane & 15;
    size_t eblk = (size_t)blockIdx.x * (16 * EDGE_NT);

    // stage one 16x128 fp32 tile (512 x 16B chunks, 2 per thread) into buffer b
    auto stage = [&](int b, int tile) {
#pragma unroll
        for (int c = t; c < 512; c += 256) {
            int row  = c >> 5;
            int col4 = (c & 31) * 4;
            const float* g = edge_attr + (eblk + (size_t)tile * 16 + row) * DD + col4;
            float* l = &sF[b][row][col4];
#if HAS_ASYNC_LDS
            __builtin_amdgcn_global_load_async_to_lds_b128(
                (gv4i*)(float*)g, (lv4i*)l, 0, 0);
#else
            *(float4*)l = *(const float4*)g;
#endif
        }
    };

    stage(0, 0);
    for (int tile = 0; tile < EDGE_NT; ++tile) {
        int b = tile & 1;
        if (tile + 1 < EDGE_NT) {
            stage(b ^ 1, tile + 1);
#if HAS_ASYNC_LDS
            __builtin_amdgcn_s_wait_asynccnt(2);
#endif
        } else {
#if HAS_ASYNC_LDS
            __builtin_amdgcn_s_wait_asynccnt(0);
#endif
        }
        __syncthreads();  // all waves' tile data visible

        v8f acc = {};
#pragma unroll
        for (int k0 = 0; k0 < DD; k0 += 32) {
            v16bf a;
#pragma unroll
            for (int j = 0; j < 8; ++j) {
                int k = k0 + ((j < 4) ? (half * 8 + 2 * j) : (half * 8 + 16 + 2 * (j - 4)));
                a[2 * j]     = (bf16)sF[b][nl][k];
                a[2 * j + 1] = (bf16)sF[b][nl][k + 1];
            }
            v16bf bb = load_b_frag(WeT, w * 16, DD, k0, lane);
            acc = __builtin_amdgcn_wmma_f32_16x16x32_bf16(false, a, false, bb, (short)0, acc, false, false);
        }

        int col = w * 16 + nl;  // head w, channel nl
        size_t e0 = eblk + (size_t)tile * 16;
#pragma unroll
        for (int r = 0; r < 8; ++r) {
            int m = r + 8 * half;
            size_t eidx = e0 + m;
            long long s = ei[eidx];
            long long d = ei[(size_t)EE + eidx];
            float ev = acc[r];
            e_bf[eidx * DD + col] = (bf16)ev;
            float kk = Kn[(size_t)s * DD + col];
            float qq = Qn[(size_t)d * DD + col];
            float p = qq * (kk + ev);
            p += __shfl_xor(p, 1, 32);
            p += __shfl_xor(p, 2, 32);
            p += __shfl_xor(p, 4, 32);
            p += __shfl_xor(p, 8, 32);
            if (nl == 0) {
                float lg = p * 0.25f;  // 1/sqrt(C), C=16
                logits[eidx * HH + w] = lg;
                atomicMax(&mmax[(size_t)d * HH + w], enc_f32(lg));
            }
        }
        __syncthreads();  // done reading buffer b before it is restaged
    }
}

// ---------- softmax numerator + denominator ----------
__global__ void k_softmax(const long long* __restrict__ ei, float* __restrict__ logits,
                          const unsigned* __restrict__ mmax, float* __restrict__ z) {
    long idx = (long)blockIdx.x * blockDim.x + threadIdx.x;  // over E*H
    long e = idx >> 3;
    int hh = (int)(idx & 7);
    long long d = ei[(size_t)EE + e];
    float m = dec_f32(mmax[(size_t)d * HH + hh]);
    float p = __expf(logits[idx] - m);
    logits[idx] = p;
    atomicAdd(&z[(size_t)d * HH + hh], p);
}

// ---------- message scatter: aggr[dst] += attn * (Vn[src] + e) ----------
__global__ void k_scatter(const long long* __restrict__ ei, const float* __restrict__ pbuf,
                          const float* __restrict__ z, const float* __restrict__ Vn,
                          const bf16* __restrict__ e_bf, float* __restrict__ aggr) {
    long idx = (long)blockIdx.x * blockDim.x + threadIdx.x;  // over E*D
    long e = idx >> 7;
    int c = (int)(idx & 127);
    int hh = c >> 4;
    long long s = ei[e];
    long long d = ei[(size_t)EE + e];
    float p  = pbuf[e * HH + hh];
    float zz = z[(size_t)d * HH + hh];
    float attn = p / (zz + 1e-16f);
    float v = Vn[(size_t)s * DD + c] + (float)e_bf[(size_t)e * DD + c];
    atomicAdd(&aggr[(size_t)d * DD + c], attn * v);
}

// ---------- host ----------
extern "C" void kernel_launch(void* const* d_in, const int* in_sizes, int n_in,
                              void* d_out, int out_size, void* d_ws, size_t ws_size,
                              hipStream_t stream) {
    const float*     x      = (const float*)d_in[0];
    const float*     eattr  = (const float*)d_in[1];
    const long long* ei     = (const long long*)d_in[2];
    const float* Wq = (const float*)d_in[3];  const float* bq = (const float*)d_in[4];
    const float* Wk = (const float*)d_in[5];  const float* bk = (const float*)d_in[6];
    const float* Wv = (const float*)d_in[7];  const float* bv = (const float*)d_in[8];
    const float* We = (const float*)d_in[9];
    const float* Ws = (const float*)d_in[10]; const float* bs = (const float*)d_in[11];
    const float* W1 = (const float*)d_in[12]; const float* b1 = (const float*)d_in[13];
    const float* W2 = (const float*)d_in[14]; const float* b2 = (const float*)d_in[15];
    const float* g1 = (const float*)d_in[16]; const float* be1 = (const float*)d_in[17];
    const float* g2 = (const float*)d_in[18]; const float* be2 = (const float*)d_in[19];
    const float* alpha = (const float*)d_in[20];

    char* ws = (char*)d_ws;
    auto alloc = [&](size_t bytes) -> char* {
        char* p = ws;
        ws += (bytes + 255) & ~(size_t)255;
        return p;
    };
    bf16* wqT = (bf16*)alloc((size_t)DD * DD * 2);
    bf16* wkT = (bf16*)alloc((size_t)DD * DD * 2);
    bf16* wvT = (bf16*)alloc((size_t)DD * DD * 2);
    bf16* weT = (bf16*)alloc((size_t)DD * DD * 2);
    bf16* wsT = (bf16*)alloc((size_t)DD * DD * 2);
    bf16* w1T = (bf16*)alloc((size_t)DD * DFF * 2);
    bf16* w2T = (bf16*)alloc((size_t)DFF * DD * 2);
    bf16*  h_bf  = (bf16*)alloc((size_t)NN * DD * 2);
    float* Qn    = (float*)alloc((size_t)NN * DD * 4);
    float* Kn    = (float*)alloc((size_t)NN * DD * 4);
    float* Vn    = (float*)alloc((size_t)NN * DD * 4);
    float* Sk    = (float*)alloc((size_t)NN * DD * 4);
    bf16*  e_bf  = (bf16*)alloc((size_t)EE * DD * 2);
    float* lgts  = (float*)alloc((size_t)EE * HH * 4);
    unsigned* mmax = (unsigned*)alloc((size_t)NN * HH * 4);
    float* z     = (float*)alloc((size_t)NN * HH * 4);
    float* aggr  = (float*)alloc((size_t)NN * DD * 4);
    float* x2    = (float*)alloc((size_t)NN * DD * 4);
    bf16*  h2_bf = (bf16*)alloc((size_t)NN * DD * 2);
    bf16*  f_bf  = (bf16*)alloc((size_t)NN * DFF * 2);

    // 1) weights -> bf16 transposed
    k_wt<<<64, 256, 0, stream>>>(Wq, wqT, DD, DD);
    k_wt<<<64, 256, 0, stream>>>(Wk, wkT, DD, DD);
    k_wt<<<64, 256, 0, stream>>>(Wv, wvT, DD, DD);
    k_wt<<<64, 256, 0, stream>>>(We, weT, DD, DD);
    k_wt<<<64, 256, 0, stream>>>(Ws, wsT, DD, DD);
    k_wt<<<256, 256, 0, stream>>>(W1, w1T, DD, DFF);
    k_wt<<<256, 256, 0, stream>>>(W2, w2T, DFF, DD);

    // 2) init atomics scratch
    k_init<<<(NN * DD) / 256, 256, 0, stream>>>(aggr, z, mmax);

    // 3) LN1
    k_ln<<<NN / 8, 256, 0, stream>>>(x, nullptr, nullptr, nullptr, g1, be1, nullptr, h_bf);

    // 4) node GEMMs (bf16 WMMA, fp32 acc)
    k_gemm<<<dim3(NN / 16, 1), 256, 0, stream>>>(h_bf, wqT, bq, nullptr, Qn, nullptr, DD, DD, 0);
    k_gemm<<<dim3(NN / 16, 1), 256, 0, stream>>>(h_bf, wkT, bk, nullptr, Kn, nullptr, DD, DD, 0);
    k_gemm<<<dim3(NN / 16, 1), 256, 0, stream>>>(h_bf, wvT, bv, nullptr, Vn, nullptr, DD, DD, 0);
    k_gemm<<<dim3(NN / 16, 1), 256, 0, stream>>>(h_bf, wsT, bs, nullptr, Sk, nullptr, DD, DD, 0);

    // 5) fused edge GEMM + logits + segment max (async double-buffered LDS pipeline)
    k_edge<<<EE / (16 * EDGE_NT), 256, 0, stream>>>(eattr, ei, weT, Qn, Kn, e_bf, lgts, mmax);

    // 6) softmax p/z, then message scatter
    k_softmax<<<(EE * HH) / 256, 256, 0, stream>>>(ei, lgts, mmax, z);
    k_scatter<<<((size_t)EE * DD) / 256, 256, 0, stream>>>(ei, lgts, z, Vn, e_bf, aggr);

    // 7) residual combine + LN2
    k_ln<<<NN / 8, 256, 0, stream>>>(x, aggr, Sk, alpha, g2, be2, x2, h2_bf);

    // 8) FFN
    k_gemm<<<dim3(NN / 16, DFF / 128), 256, 0, stream>>>(h2_bf, w1T, b1, nullptr, nullptr, f_bf, DD, DFF, 1);
    k_gemm<<<dim3(NN / 16, 1), 256, 0, stream>>>(f_bf, w2T, b2, x2, (float*)d_out, nullptr, DFF, DD, 2);
}
